// JointSentenceBiLSTM1_81114752352619
// MI455X (gfx1250) — matrix operand order repaired and checked by reading, not compiled, not run on hardware
//
#include <hip/hip_runtime.h>
#include <hip/hip_bf16.h>
#include <math.h>

// Sizes fixed by the reference: B=32, L=128, D=300(->320), H=128, 4H=512,
// E=34(->48), A=36(->48), 2H=256, E-1=33.

typedef _Float16 half_t;
typedef __attribute__((ext_vector_type(16))) _Float16 v16h;
typedef __attribute__((ext_vector_type(8)))  _Float16 v8h;
typedef __attribute__((ext_vector_type(8)))  float    v8f;

#define DEVINL __device__ __forceinline__

DEVINL v8f wmma_acc(v16h a, v16h b, v8f c) {
  // D = A(16x32) * B(32x16) + C, f16 in / f32 acc
  return __builtin_amdgcn_wmma_f32_16x16x32_f16(false, a, false, b, (short)0, c,
                                                false, false);
}

// A-matrix fragment (16x32, row-major, lda halves). Lane holds
// M = lane&15, K values h*16 + 0..15 (h = lane>>4):
// two contiguous 16-byte loads.
DEVINL v16h load_a16(const half_t* tile, int lda, int lane) {
  const int m = lane & 15, h = lane >> 4;
  const half_t* p = tile + (size_t)m * lda + h * 16;
  v8h lo = *(const v8h*)(p);
  v8h hi = *(const v8h*)(p + 8);
  v16h r;
#pragma unroll
  for (int e = 0; e < 8; ++e) { r[e] = lo[e]; r[e + 8] = hi[e]; }
  return r;
}

// B-matrix fragment (32x16) read from row-major weights W[N][K] so that
// B[k][n] = W[col0+n][k0+k]. Lane holds N = lane&15, K = h*16 + 0..15:
// one contiguous 32-byte load.
DEVINL v16h load_b16(const half_t* W, int ldw, int col0, int k0, int lane) {
  const int n = lane & 15, h = lane >> 4;
  return *(const v16h*)(W + (size_t)(col0 + n) * ldw + k0 + h * 16);
}

// ---------------------------------------------------------------- prep ----

// Gather embedding rows, convert to f16, zero-pad D 300 -> 320.
__global__ void k_embed(const int* __restrict__ ids, const float* __restrict__ emb,
                        half_t* __restrict__ xbuf) {
  const int row = blockIdx.x;               // b*L + l, 0..4095
  const int id = ids[row];
  const float* src = emb + (size_t)id * 300;
  half_t* dst = xbuf + (size_t)row * 320;
  for (int c = threadIdx.x; c < 320; c += blockDim.x)
    dst[c] = (c < 300) ? (half_t)src[c] : (half_t)0.0f;
}

// Convert f32 weight block [rows, cols] (row stride src_ld, col offset
// col_off) into zero-padded f16 [gridDim.x, dst_ld].
__global__ void k_pack(const float* __restrict__ src, int src_ld, int col_off,
                       int rows, int cols, half_t* __restrict__ dst, int dst_ld) {
  const int r = blockIdx.x;
  half_t* d = dst + (size_t)r * dst_ld;
  for (int c = threadIdx.x; c < dst_ld; c += blockDim.x) {
    float v = (r < rows && c < cols) ? src[(size_t)r * src_ld + col_off + c] : 0.0f;
    d[c] = (half_t)v;
  }
}

// Combined biases + padded head biases + Wg (= arg_w[:, 512:545], f32).
__global__ void k_misc(const float* __restrict__ bif, const float* __restrict__ bhf,
                       const float* __restrict__ bib, const float* __restrict__ bhb,
                       const float* __restrict__ event_b, const float* __restrict__ arg_b,
                       const float* __restrict__ arg_w,
                       float* __restrict__ bias_lstm, float* __restrict__ evb,
                       float* __restrict__ agb, float* __restrict__ wg) {
  const int t = blockIdx.x * blockDim.x + threadIdx.x;
  if (t < 512)        bias_lstm[t] = bif[t] + bhf[t];
  else if (t < 1024)  bias_lstm[t] = bib[t - 512] + bhb[t - 512];
  if (t < 48) evb[t] = (t < 34) ? event_b[t] : -1e30f;
  if (t < 48) agb[t] = (t < 36) ? arg_b[t] : 0.0f;
  if (t < 36 * 33) wg[t] = arg_w[(size_t)(t / 33) * 545 + 512 + (t % 33)];
}

// ---------------------------------------------------------------- GEMM ----

// C[M,N](f32) = A[M,K](f16) @ W[N,K](f16).T (+ bias). 8 waves/block, one
// 16x16 output tile per wave. grid = (M/16, ceil(N/128)).
__global__ void k_gemm(const half_t* __restrict__ A, int lda,
                       const half_t* __restrict__ W, int ldw,
                       const float* __restrict__ bias,
                       float* __restrict__ C, int ldc, int N, int K) {
  const int lane = threadIdx.x & 31;
  const int wave = threadIdx.x >> 5;
  const int n0 = blockIdx.y * 128 + wave * 16;
  if (n0 >= N) return;                       // wave-uniform: EXEC stays all-1s
  const half_t* Ap = A + (size_t)blockIdx.x * 16 * lda;
  v8f acc = {};
  for (int k = 0; k < K; k += 32) {
    if (k + 32 < K) __builtin_prefetch(Ap + k + 32, 0, 3);
    v16h a = load_a16(Ap + k, lda, lane);
    v16h b = load_b16(W, ldw, n0, k, lane);
    acc = wmma_acc(a, b, acc);
  }
  const int n = lane & 15, h = lane >> 4;
  const float bv = bias ? bias[n0 + n] : 0.0f;
  const size_t base = ((size_t)blockIdx.x * 16 + h * 8) * ldc + n0 + n;
#pragma unroll
  for (int r = 0; r < 8; ++r) C[base + (size_t)r * ldc] = acc[r] + bv;
}

// Input-projection GEMM with batch-minor transposed store:
// CT[t][col][b] = (x @ w_ih.T)[b*128+t][col], CT shape [128][512][32] f32.
// Row tiles never straddle a batch boundary (128 % 16 == 0).
__global__ void k_gemm_xpT(const half_t* __restrict__ A, int lda,
                           const half_t* __restrict__ W, int ldw,
                           float* __restrict__ CT, int K) {
  const int lane = threadIdx.x & 31;
  const int wave = threadIdx.x >> 5;
  const int n0 = blockIdx.y * 128 + wave * 16;
  const half_t* Ap = A + (size_t)blockIdx.x * 16 * lda;
  v8f acc = {};
  for (int k = 0; k < K; k += 32) {
    if (k + 32 < K) __builtin_prefetch(Ap + k + 32, 0, 3);
    v16h a = load_a16(Ap + k, lda, lane);
    v16h b = load_b16(W, ldw, n0, k, lane);
    acc = wmma_acc(a, b, acc);
  }
  const int n = lane & 15, h = lane >> 4;
  const int row0 = blockIdx.x * 16 + h * 8;
#pragma unroll
  for (int r = 0; r < 8; ++r) {
    const int row = row0 + r;                // b*128 + t
    const int b = row >> 7, t = row & 127;
    CT[(((size_t)t * 512) + n0 + n) * 32 + b] = acc[r];
  }
}

// ---------------------------------------------------------------- LSTM ----

// One workgroup per direction (grid=2), 256 threads = 8 waves. Wave w owns
// hidden-column slice [16w,16w+16) of EACH gate block, so all four gate
// accumulators for a given (batch, col) sit in the same lane -> gate math is
// pure register VALU. h lives in LDS (f16), c in registers, and all 16
// w_hh B-fragments (4 gates x 4 k-steps = 128 VGPRs) are hoisted into
// registers for the whole recurrence: the serial loop touches no weight
// memory. z-init reads xprojT [t][col][b]: one aligned v8f per (gate,mtile).
__global__ void k_lstm(const half_t* __restrict__ whh_all,
                       const float* __restrict__ xprojT_all, /* [2][128][512][32] */
                       const float* __restrict__ bias_all,
                       half_t* __restrict__ hidden /* [4096][256] f16 */) {
  const int dir  = blockIdx.x;
  const int lane = threadIdx.x & 31;
  const int wave = threadIdx.x >> 5;
  const int n0 = wave * 16;
  const int n  = n0 + (lane & 15);
  const int mh = (lane >> 4) * 8;
  const half_t* whh = whh_all + (size_t)dir * 512 * 128;
  const float*  xpt = xprojT_all + (size_t)dir * 128 * 512 * 32;
  const float*  bi  = bias_all + dir * 512;

  __shared__ __align__(32) half_t hs[32][128];   // h state, all batches
  for (int i = threadIdx.x; i < 32 * 128; i += blockDim.x)
    (&hs[0][0])[i] = (half_t)0.0f;

  float bg[4];
#pragma unroll
  for (int g = 0; g < 4; ++g) bg[g] = bi[g * 128 + n];

  v16h wfrag[4][4];                               // resident w_hh fragments
#pragma unroll
  for (int g = 0; g < 4; ++g)
#pragma unroll
    for (int kk = 0; kk < 4; ++kk)
      wfrag[g][kk] = load_b16(whh, 128, g * 128 + n0, kk * 32, lane);

  float cst[2][8];                                // c state (registers)
#pragma unroll
  for (int mt = 0; mt < 2; ++mt)
#pragma unroll
    for (int r = 0; r < 8; ++r) cst[mt][r] = 0.0f;

  __syncthreads();
  for (int tt = 0; tt < 128; ++tt) {
    const int t = dir ? (127 - tt) : tt;
    v8f acc[8];                                   // [gate*2 + mtile]
#pragma unroll
    for (int g = 0; g < 4; ++g)
#pragma unroll
      for (int mt = 0; mt < 2; ++mt) {
        // 8 consecutive batches at (t, col g*128+n): one 32B-aligned load.
        v8f z = *(const v8f*)(xpt + (((size_t)t * 512) + g * 128 + n) * 32 +
                              mt * 16 + mh);
#pragma unroll
        for (int r = 0; r < 8; ++r) z[r] += bg[g];
        acc[g * 2 + mt] = z;
      }
#pragma unroll
    for (int kk = 0; kk < 4; ++kk) {
      v16h a0 = load_a16(&hs[0][0]  + kk * 32, 128, lane);  // batches 0..15
      v16h a1 = load_a16(&hs[16][0] + kk * 32, 128, lane);  // batches 16..31
#pragma unroll
      for (int g = 0; g < 4; ++g) {
        acc[g * 2 + 0] = wmma_acc(a0, wfrag[g][kk], acc[g * 2 + 0]);
        acc[g * 2 + 1] = wmma_acc(a1, wfrag[g][kk], acc[g * 2 + 1]);
      }
    }
    __syncthreads();                              // all reads of hs done
#pragma unroll
    for (int mt = 0; mt < 2; ++mt)
#pragma unroll
      for (int r = 0; r < 8; ++r) {
        const float iv = 1.0f / (1.0f + __expf(-acc[0 * 2 + mt][r]));
        const float fv = 1.0f / (1.0f + __expf(-acc[1 * 2 + mt][r]));
        const float gv = tanhf(acc[2 * 2 + mt][r]);
        const float ov = 1.0f / (1.0f + __expf(-acc[3 * 2 + mt][r]));
        const float cn = fv * cst[mt][r] + iv * gv;
        cst[mt][r] = cn;
        const float hv = ov * tanhf(cn);
        const int b = mt * 16 + mh + r;
        const half_t hh = (half_t)hv;
        hs[b][n] = hh;
        hidden[((size_t)b * 128 + t) * 256 + dir * 128 + n] = hh;
      }
    __syncthreads();                              // hs updated for next t
  }
}

// ---------------------------------------------------------------- heads ----

// Event argmax + copy event logits to output.
__global__ void k_evpred(const float* __restrict__ evl, float* __restrict__ out_ev,
                         int* __restrict__ ev_pred) {
  const int row = blockIdx.x * blockDim.x + threadIdx.x;
  if (row >= 4096) return;
  const float* p = evl + (size_t)row * 48;
  float best = p[0];
  int bi = 0;
#pragma unroll
  for (int a = 0; a < 34; ++a) {
    const float v = p[a];
    out_ev[(size_t)row * 34 + a] = v;
    if (v > best) { best = v; bi = a; }           // first max wins (strict >)
  }
  ev_pred[row] = bi;
}

// Sequential g-scan: independent per batch element -> one block per b.
// arg_logit[i][l][a] = base[l][a] + trigproj[i][a] + g[l] . Wg[a].
// base is cached in LDS for the whole chain.
__global__ void k_scan(const float* __restrict__ base, const float* __restrict__ trig,
                       const int* __restrict__ ev_pred, const float* __restrict__ wg,
                       float* __restrict__ out_arg) {
  const int b = blockIdx.x;
  const int tid = threadIdx.x;
  __shared__ float g[128][33];
  __shared__ float wgs[36][33];
  __shared__ float base_s[128][36];
  __shared__ float logits[128][36];
  __shared__ float trig_s[36];
  __shared__ int ev_s;
  for (int i = tid; i < 128 * 33; i += blockDim.x) (&g[0][0])[i] = 0.0f;
  for (int i = tid; i < 36 * 33; i += blockDim.x) (&wgs[0][0])[i] = wg[i];
  for (int i = tid; i < 128 * 36; i += blockDim.x)
    (&base_s[0][0])[i] = base[((size_t)b * 128 + (i / 36)) * 48 + (i % 36)];
  __syncthreads();
  for (int i = 0; i < 128; ++i) {
    if (tid < 36) trig_s[tid] = trig[((size_t)b * 128 + i) * 48 + tid];
    if (tid == 0) ev_s = ev_pred[b * 128 + i];
    __syncthreads();
    for (int idx = tid; idx < 128 * 36; idx += blockDim.x) {
      const int l = idx / 36, a = idx % 36;
      float s = base_s[l][a] + trig_s[a];
#pragma unroll
      for (int j = 0; j < 33; ++j) s += g[l][j] * wgs[a][j];
      logits[l][a] = s;
      out_arg[(((size_t)b * 128 + i) * 128 + l) * 36 + a] = s;
    }
    __syncthreads();
    if (tid < 128) {
      const float* lp = logits[tid];
      float best = lp[0];
      int am = 0;
      for (int a = 1; a < 36; ++a)
        if (lp[a] > best) { best = lp[a]; am = a; }
      if (ev_s > 0 && am > 0) g[tid][ev_s - 1] = 1.0f;
    }
    __syncthreads();
  }
}

// ---------------------------------------------------------------- host ----

extern "C" void kernel_launch(void* const* d_in, const int* in_sizes, int n_in,
                              void* d_out, int out_size, void* d_ws, size_t ws_size,
                              hipStream_t stream) {
  const int*   ids     = (const int*)d_in[0];
  const float* emb     = (const float*)d_in[1];
  const float* w_ih_f  = (const float*)d_in[2];
  const float* w_hh_f  = (const float*)d_in[3];
  const float* b_ih_f  = (const float*)d_in[4];
  const float* b_hh_f  = (const float*)d_in[5];
  const float* w_ih_b  = (const float*)d_in[6];
  const float* w_hh_b  = (const float*)d_in[7];
  const float* b_ih_b  = (const float*)d_in[8];
  const float* b_hh_b  = (const float*)d_in[9];
  const float* event_w = (const float*)d_in[10];
  const float* event_b = (const float*)d_in[11];
  const float* arg_w   = (const float*)d_in[12];
  const float* arg_b   = (const float*)d_in[13];
  float* out = (float*)d_out;
  (void)in_sizes; (void)n_in; (void)out_size; (void)ws_size;

  char* ws = (char*)d_ws;
  size_t off = 0;
  auto alloc = [&](size_t bytes) {
    char* p = ws + off;
    off = (off + bytes + 255) & ~(size_t)255;
    return p;
  };
  half_t* xbuf   = (half_t*)alloc((size_t)4096 * 320 * 2);
  half_t* wih16  = (half_t*)alloc((size_t)2 * 512 * 320 * 2);
  half_t* whh16  = (half_t*)alloc((size_t)2 * 512 * 128 * 2);
  float*  biasl  = (float*)alloc((size_t)2 * 512 * 4);
  float*  xprojT = (float*)alloc((size_t)2 * 128 * 512 * 32 * 4);
  half_t* hidden = (half_t*)alloc((size_t)4096 * 256 * 2);
  half_t* evw16  = (half_t*)alloc((size_t)48 * 256 * 2);
  half_t* wb16   = (half_t*)alloc((size_t)48 * 256 * 2);
  half_t* wt16   = (half_t*)alloc((size_t)48 * 256 * 2);
  float*  evb    = (float*)alloc(48 * 4);
  float*  agb    = (float*)alloc(48 * 4);
  float*  wg     = (float*)alloc(36 * 33 * 4);
  float*  evl    = (float*)alloc((size_t)4096 * 48 * 4);
  float*  basep  = (float*)alloc((size_t)4096 * 48 * 4);
  float*  trigp  = (float*)alloc((size_t)4096 * 48 * 4);
  int*    evpred = (int*)alloc((size_t)4096 * 4);

  // Stage 0: gather/convert/pad.
  k_embed<<<4096, 64, 0, stream>>>(ids, emb, xbuf);
  k_pack<<<512, 128, 0, stream>>>(w_ih_f, 300, 0, 512, 300, wih16, 320);
  k_pack<<<512, 128, 0, stream>>>(w_ih_b, 300, 0, 512, 300, wih16 + 512 * 320, 320);
  k_pack<<<512, 128, 0, stream>>>(w_hh_f, 128, 0, 512, 128, whh16, 128);
  k_pack<<<512, 128, 0, stream>>>(w_hh_b, 128, 0, 512, 128, whh16 + 512 * 128, 128);
  k_pack<<<48, 128, 0, stream>>>(event_w, 256, 0, 34, 256, evw16, 256);
  k_pack<<<48, 128, 0, stream>>>(arg_w, 545, 0, 36, 256, wb16, 256);
  k_pack<<<48, 128, 0, stream>>>(arg_w, 545, 256, 36, 256, wt16, 256);
  k_misc<<<2, 1024, 0, stream>>>(b_ih_f, b_hh_f, b_ih_b, b_hh_b, event_b, arg_b,
                                 arg_w, biasl, evb, agb, wg);

  // Stage 1: input projections -> batch-minor layout [t][col][b].
  dim3 gx(256, 4);
  k_gemm_xpT<<<gx, 256, 0, stream>>>(xbuf, 320, wih16, 320, xprojT, 320);
  k_gemm_xpT<<<gx, 256, 0, stream>>>(xbuf, 320, wih16 + 512 * 320, 320,
                                     xprojT + (size_t)128 * 512 * 32, 320);

  // Stage 2: bidirectional LSTM recurrence -> hidden [4096,256] f16.
  k_lstm<<<2, 256, 0, stream>>>(whh16, xprojT, biasl, hidden);

  // Stage 3: head GEMMs (event logits, trigger-fixed and trigger-varying
  // parts of the argument head), all precomputed once.
  dim3 gh(256, 1);
  k_gemm<<<gh, 256, 0, stream>>>(hidden, 256, evw16, 256, evb, evl, 48, 48, 256);
  k_gemm<<<gh, 256, 0, stream>>>(hidden, 256, wb16, 256, agb, basep, 48, 48, 256);
  k_gemm<<<gh, 256, 0, stream>>>(hidden, 256, wt16, 256, nullptr, trigp, 48, 48, 256);

  // Stage 4: event argmax + event-logit output.
  k_evpred<<<32, 128, 0, stream>>>(evl, out, evpred);

  // Stage 5: sequential g-scan, one chain per batch element.
  k_scan<<<32, 256, 0, stream>>>(basep, trigp, evpred, wg, out + (size_t)4096 * 34);
}
